// WindowAttention_43679817400866
// MI455X (gfx1250) — compile-verified, matrix operand order checked by reading, not run on hardware
//
#include <hip/hip_runtime.h>

typedef __bf16 bf16;
typedef __attribute__((ext_vector_type(16))) __bf16 bf16x16;
typedef __attribute__((ext_vector_type(8)))  float  f32x8;

#define WIN_N   49
#define DIM     512
#define HEADS   16
#define HDIM    32
#define BW      2048
#define GK      512
#define M_TOTAL (BW * WIN_N) /* 100352 = 392 * 256 */

__device__ __forceinline__ unsigned short bfbits(float x) {
  bf16 b = (bf16)x;                       // native f32->bf16 (RTE) on gfx1250
  unsigned short u; __builtin_memcpy(&u, &b, 2); return u;
}
__device__ __forceinline__ bf16 f2bf(float x) { return (bf16)x; }
__device__ __forceinline__ void store_out(bf16* p, float v)  { *p = f2bf(v); }
__device__ __forceinline__ void store_out(float* p, float v) { *p = v; }

// gfx1250 16-bit WMMA fragment K-swizzle (ISA 7.12.2), closed form:
// half-lane group = (k>>3)&1 ; lane element index = (k&7) | ((k>>1)&8)
__device__ __forceinline__ int khalf(int k) { return (k >> 3) & 1; }
__device__ __forceinline__ int kelem(int k) { return (k & 7) | ((k >> 1) & 8); }

// Fragment-ready LDS layout: [tile][lane(32)][16 bf16] -> one 32B vector load.
__device__ __forceinline__ bf16x16 frag_ld(const bf16* s, int tile, int lane) {
  return *reinterpret_cast<const bf16x16*>(s + ((tile * 32 + lane) << 4));
}

__device__ __forceinline__ f32x8 wmma_bf16(bf16x16 a, bf16x16 b, f32x8 c) {
  return __builtin_amdgcn_wmma_f32_16x16x32_bf16(false, a, false, b, (short)0, c,
                                                 false, false);
}

// ---------------------------------------------------------------------------
// Bias table with key-mask baked in: bias_pad[h][m][n] (64x64):
//   n>=49 -> -1e30 (masks padded keys, so softmax needs no cndmask at all)
// ---------------------------------------------------------------------------
__global__ __launch_bounds__(256) void bias_pad_kernel(
    const float* __restrict__ rpb, const int* __restrict__ rel,
    float* __restrict__ bias_pad) {
  const int e = blockIdx.x * 256 + threadIdx.x;   // 16*64*64 = 65536 total
  const int h = e >> 12, mn = e & 4095, m = mn >> 6, n = mn & 63;
  float v;
  if (n >= WIN_N)      v = -1e30f;
  else if (m < WIN_N)  v = rpb[rel[m * WIN_N + n] * HEADS + h];
  else                 v = 0.f;
  bias_pad[e] = v;
}

// ---------------------------------------------------------------------------
// GEMM: C[M,NC] = A[M,512] * B[512,NC] + bias.
// 256x128 macro-tile, 256 thr = 8 waves (4M x 2N), wave tile 64x64 (16 WMMA
// per K-step vs 8 fragment loads), K-step 32, double-buffered LDS with a
// single barrier per K-step, reg->LDS software pipeline for global loads.
// ---------------------------------------------------------------------------
template <typename AT, typename OT, bool QKV_SCALE, int NC>
__global__ __launch_bounds__(256) void gemm_bias_kernel(
    const AT* __restrict__ A, const float* __restrict__ B,
    const float* __restrict__ bias, OT* __restrict__ C) {
  __shared__ __align__(32) bf16 As[2][16 * 32 * 16];  // 16 M-tiles x2 buffers
  __shared__ __align__(32) bf16 Bs[2][8 * 32 * 16];   //  8 N-tiles x2 buffers

  const int tid = threadIdx.x, lane = tid & 31, w = tid >> 5;
  const int mb = blockIdx.x * 256, nb = blockIdx.y * 128;
  const int mt0 = (w & 3) * 4, nt0 = (w >> 2) * 4;

  f32x8 acc[4][4];
#pragma unroll
  for (int mi = 0; mi < 4; ++mi)
#pragma unroll
    for (int ni = 0; ni < 4; ++ni)
      acc[mi][ni] = f32x8{0.f, 0.f, 0.f, 0.f, 0.f, 0.f, 0.f, 0.f};

  float4 bR[4];
  float4 aRf[8];  // AT == float path
  uint4  aRh[4];  // AT == bf16 path (no conversion needed)

  auto load_tile = [&](int kk) {
#pragma unroll
    for (int p = 0; p < 4; ++p) {
      const int e = tid + 256 * p;  // B: 32k x 128n, float4 along n
      bR[p] = *reinterpret_cast<const float4*>(
          &B[(size_t)(kk + (e >> 5)) * NC + nb + (e & 31) * 4]);
    }
    if constexpr (sizeof(AT) == 4) {
#pragma unroll
      for (int p = 0; p < 8; ++p) {
        const int e = tid + 256 * p;  // A: 256m x 32k, float4 along k
        aRf[p] = *reinterpret_cast<const float4*>(
            &A[(size_t)(mb + (e >> 3)) * GK + kk + (e & 7) * 4]);
      }
    } else {
#pragma unroll
      for (int p = 0; p < 4; ++p) {
        const int e = tid + 256 * p;  // A: 256m x 32k, 8 bf16 along k
        aRh[p] = *reinterpret_cast<const uint4*>(
            &A[(size_t)(mb + (e >> 2)) * GK + kk + (e & 3) * 8]);
      }
    }
  };
  auto store_tile = [&](int c) {
#pragma unroll
    for (int p = 0; p < 4; ++p) {
      const int e = tid + 256 * p;
      const int br = e >> 5, bc4 = (e & 31) * 4;  // same k, 4 consecutive n
      const int base = ((bc4 >> 4) * 32) + (bc4 & 15) + 16 * khalf(br);
      const int kel = kelem(br);
      Bs[c][(base + 0) * 16 + kel] = f2bf(bR[p].x);
      Bs[c][(base + 1) * 16 + kel] = f2bf(bR[p].y);
      Bs[c][(base + 2) * 16 + kel] = f2bf(bR[p].z);
      Bs[c][(base + 3) * 16 + kel] = f2bf(bR[p].w);
    }
    if constexpr (sizeof(AT) == 4) {
#pragma unroll
      for (int p = 0; p < 8; ++p) {
        const int e = tid + 256 * p;
        const int ar = e >> 3, ak4 = (e & 7) * 4;  // 4 consecutive k -> 4
        const int idx =                            // consecutive frag elems
            (((ar >> 4) * 32) + (ar & 15) + 16 * khalf(ak4)) * 16 + kelem(ak4);
        uint2 u;
        u.x = (unsigned)bfbits(aRf[p].x) | ((unsigned)bfbits(aRf[p].y) << 16);
        u.y = (unsigned)bfbits(aRf[p].z) | ((unsigned)bfbits(aRf[p].w) << 16);
        *reinterpret_cast<uint2*>(&As[c][idx]) = u;
      }
    } else {
#pragma unroll
      for (int p = 0; p < 4; ++p) {
        const int e = tid + 256 * p;
        const int ar = e >> 2, ak8 = (e & 3) * 8;  // 8 consecutive k -> 16B
        const int idx =
            (((ar >> 4) * 32) + (ar & 15) + 16 * khalf(ak8)) * 16 + kelem(ak8);
        *reinterpret_cast<uint4*>(&As[c][idx]) = aRh[p];
      }
    }
  };

  load_tile(0);
  store_tile(0);
  int c = 0;
  for (int kk = 0; kk < GK; kk += 32, c ^= 1) {
    __syncthreads();  // buffer c fully staged; buffer c^1 free to overwrite
    const bool more = (kk + 32) < GK;
    if (more) load_tile(kk + 32);  // HBM latency hides under the WMMAs
    bf16x16 af[4], bfr[4];
#pragma unroll
    for (int mi = 0; mi < 4; ++mi) af[mi] = frag_ld(As[c], mt0 + mi, lane);
#pragma unroll
    for (int ni = 0; ni < 4; ++ni) bfr[ni] = frag_ld(Bs[c], nt0 + ni, lane);
#pragma unroll
    for (int mi = 0; mi < 4; ++mi)
#pragma unroll
      for (int ni = 0; ni < 4; ++ni)
        acc[mi][ni] = wmma_bf16(af[mi], bfr[ni], acc[mi][ni]);
    if (more) store_tile(c ^ 1);  // write other buffer; no barrier needed here
  }

  const int half = lane >> 4, nl = lane & 15;
  const float sc = (QKV_SCALE && nb < DIM) ? 0.17677669529663687f : 1.0f;
#pragma unroll
  for (int mi = 0; mi < 4; ++mi)
#pragma unroll
    for (int ni = 0; ni < 4; ++ni)
#pragma unroll
      for (int r = 0; r < 8; ++r) {
        const int row = mb + (mt0 + mi) * 16 + r + 8 * half;
        const int col = nb + (nt0 + ni) * 16 + nl;
        store_out(&C[(size_t)row * NC + col], (acc[mi][ni][r] + bias[col]) * sc);
      }
}

// ---------------------------------------------------------------------------
// Per-(window, head) attention: S = Q K^T + bias (bias fused as WMMA C and
// carrying the key mask), softmax with no masking ops, O = P V.
// 128 threads = 4 waves, one 16-row strip each.
// ---------------------------------------------------------------------------
__global__ __launch_bounds__(128) void window_attn_kernel(
    const bf16* __restrict__ qkv, const float* __restrict__ bias_pad,
    bf16* __restrict__ attn_out) {
  __shared__ __align__(32) bf16 Qs[4 * 32 * 16];  // 4 row-tiles (A layout)
  __shared__ __align__(32) bf16 Ks[4 * 32 * 16];  // 4 row-tiles (B=K^T layout)
  __shared__ __align__(32) bf16 Vs[4 * 32 * 16];  // [kt*2+nt] k-major tiles
  __shared__ __align__(32) bf16 Ps[8 * 32 * 16];  // [mt*2+kt] A-layout tiles

  const int b = blockIdx.x >> 4, h = blockIdx.x & 15;
  const int tid = threadIdx.x, lane = tid & 31, w = tid >> 5;
  const size_t rowbase = (size_t)b * WIN_N;

  // stage Q/K/V head slices: 16B global loads, 16B LDS stores for Q/K
  const int d8 = (tid & 3) * 8;   // 8 consecutive head-dim elems
  const int tq = tid >> 2;        // token 0..31 (+32 on second pass)
#pragma unroll
  for (int j = 0; j < 2; ++j) {
    const int t = tq + 32 * j;
    uint4 q4 = {0, 0, 0, 0}, k4 = q4, v4 = q4;
    if (t < WIN_N) {
      const bf16* src = qkv + (rowbase + t) * (3 * DIM) + h * HDIM + d8;
      q4 = *reinterpret_cast<const uint4*>(src);
      k4 = *reinterpret_cast<const uint4*>(src + DIM);
      v4 = *reinterpret_cast<const uint4*>(src + 2 * DIM);
    }
    const int rm = (((t >> 4) * 32) + (t & 15) + 16 * khalf(d8)) * 16 + kelem(d8);
    *reinterpret_cast<uint4*>(&Qs[rm]) = q4;
    *reinterpret_cast<uint4*>(&Ks[rm]) = k4;
    // V as k-major B tiles: k = t, n = d8..d8+7 (scatter across lanes)
    const int kin = t & 31, kel = kelem(kin), kh = khalf(kin);
    const int vtbase = ((t >> 5) * 2 + (d8 >> 4)) * 32;
    const bf16* vp = reinterpret_cast<const bf16*>(&v4);
#pragma unroll
    for (int c = 0; c < 8; ++c)
      Vs[(vtbase + (d8 & 15) + c + 16 * kh) * 16 + kel] = vp[c];
  }
  __syncthreads();

  const int half = lane >> 4, nl = lane & 15;

  // S strip rows [16w,16w+16): bias (with key mask) preloaded into C operand
  f32x8 sfrag[4];
  const bf16x16 aq = frag_ld(Qs, w, lane);
#pragma unroll
  for (int nt = 0; nt < 4; ++nt) {
    f32x8 cb;
#pragma unroll
    for (int r = 0; r < 8; ++r)
      cb[r] = bias_pad[(h * 64 + 16 * w + r + 8 * half) * 64 + 16 * nt + nl];
    sfrag[nt] = wmma_bf16(aq, frag_ld(Ks, nt, lane), cb);
  }

  // softmax (mask already in S); one C-fragment row lives in one 16-lane half
#pragma unroll
  for (int r = 0; r < 8; ++r) {
    const int m = 16 * w + r + 8 * half;
    float v[4];
#pragma unroll
    for (int nt = 0; nt < 4; ++nt) v[nt] = sfrag[nt][r];
    float mx = fmaxf(fmaxf(v[0], v[1]), fmaxf(v[2], v[3]));
    for (int off = 1; off < 16; off <<= 1) mx = fmaxf(mx, __shfl_xor(mx, off, 32));
    float sum = 0.f;
#pragma unroll
    for (int nt = 0; nt < 4; ++nt) { v[nt] = __expf(v[nt] - mx); sum += v[nt]; }
    for (int off = 1; off < 16; off <<= 1) sum += __shfl_xor(sum, off, 32);
    const float inv = __builtin_amdgcn_rcpf(sum);
#pragma unroll
    for (int nt = 0; nt < 4; ++nt) {
      const int n = 16 * nt + nl, kin = n & 31;
      Ps[((((m >> 4) * 2 + (n >> 5)) * 32) + (m & 15) + 16 * khalf(kin)) * 16 +
         kelem(kin)] = f2bf(v[nt] * inv);
    }
  }
  __syncthreads();

  // O strip = P[16w, 0:64] @ V[0:64, 0:32]
  const bf16x16 ap0 = frag_ld(Ps, w * 2 + 0, lane);
  const bf16x16 ap1 = frag_ld(Ps, w * 2 + 1, lane);
  f32x8 o[2];
#pragma unroll
  for (int ni = 0; ni < 2; ++ni) {
    f32x8 z = f32x8{0.f, 0.f, 0.f, 0.f, 0.f, 0.f, 0.f, 0.f};
    z = wmma_bf16(ap0, frag_ld(Vs, 0 * 2 + ni, lane), z);
    o[ni] = wmma_bf16(ap1, frag_ld(Vs, 1 * 2 + ni, lane), z);
  }
#pragma unroll
  for (int ni = 0; ni < 2; ++ni)
#pragma unroll
    for (int r = 0; r < 8; ++r) {
      const int m = 16 * w + r + 8 * half;
      if (m < WIN_N)
        attn_out[(rowbase + m) * DIM + h * HDIM + 16 * ni + nl] = f2bf(o[ni][r]);
    }
}

extern "C" void kernel_launch(void* const* d_in, const int* in_sizes, int n_in,
                              void* d_out, int out_size, void* d_ws, size_t ws_size,
                              hipStream_t stream) {
  (void)in_sizes; (void)n_in; (void)out_size; (void)ws_size;
  const float* x      = (const float*)d_in[0];
  const float* qkv_w  = (const float*)d_in[1];
  const float* qkv_b  = (const float*)d_in[2];
  const float* proj_w = (const float*)d_in[3];
  const float* proj_b = (const float*)d_in[4];
  const float* rpb    = (const float*)d_in[5];
  const int*   rel    = (const int*)d_in[6];
  float* out = (float*)d_out;

  bf16*  qkv_ws   = (bf16*)d_ws;                            // [100352,1536] bf16
  bf16*  attn_ws  = qkv_ws + (size_t)M_TOTAL * 3 * DIM;     // [100352, 512] bf16
  float* bias_pad = (float*)(attn_ws + (size_t)M_TOTAL * DIM);  // [16,64,64] f32

  // 0) padded+masked bias table
  bias_pad_kernel<<<dim3((HEADS * 64 * 64) / 256), 256, 0, stream>>>(rpb, rel,
                                                                     bias_pad);
  // 1) QKV projection (+bias, Q pre-scaled), f32 -> bf16 workspace
  gemm_bias_kernel<float, bf16, true, 3 * DIM>
      <<<dim3(M_TOTAL / 256, (3 * DIM) / 128), 256, 0, stream>>>(x, qkv_w, qkv_b,
                                                                 qkv_ws);
  // 2) windowed attention per (window, head)
  window_attn_kernel<<<dim3(BW * HEADS), 128, 0, stream>>>(qkv_ws, bias_pad,
                                                           attn_ws);
  // 3) output projection (+bias) -> f32
  gemm_bias_kernel<bf16, float, false, DIM>
      <<<dim3(M_TOTAL / 256, DIM / 128), 256, 0, stream>>>(attn_ws, proj_w, proj_b,
                                                           out);
}
